// TransitionDown_56538949485052
// MI455X (gfx1250) — matrix operand order repaired and checked by reading, not compiled, not run on hardware
//
#include <hip/hip_runtime.h>

typedef __attribute__((ext_vector_type(16))) _Float16 v16h;
typedef __attribute__((ext_vector_type(8)))  _Float16 v8h;
typedef __attribute__((ext_vector_type(8)))  float    v8f;

#define B_    4
#define N_    16384
#define CIN   64
#define COUT  128
#define M_    4096
#define KNN   16
#define S_    (B_ * M_ * KNN)     /* 262144 GEMM columns */
#define BN_EPS 1e-5f

// ---------------------------------------------------------------------------
// 1) Furthest point sampling: one block per batch, sequential M iterations.
//    Per-wave shfl_xor argmax, then one cross-wave LDS round: 2 barriers/iter.
//    dist[] in global scratch (L2-resident at 64KB/batch).
// ---------------------------------------------------------------------------
__global__ void __launch_bounds__(1024)
fps_kernel(const float* __restrict__ coords, float* __restrict__ dist,
           int* __restrict__ fps_idx) {
  const int b = blockIdx.x;
  const int tid = threadIdx.x;
  const int lane = tid & 31, wv = tid >> 5;
  __shared__ float swv[32];
  __shared__ int   swi[32];
  __shared__ int   swinner;
  const float* cx = coords + (size_t)(b * 3 + 0) * N_;
  const float* cy = coords + (size_t)(b * 3 + 1) * N_;
  const float* cz = coords + (size_t)(b * 3 + 2) * N_;
  float* db = dist + (size_t)b * N_;
  for (int i = tid; i < N_; i += 1024) db[i] = 3.4e38f;
  if (tid == 0) fps_idx[b * M_] = 0;
  __syncthreads();
  int last = 0;
  for (int s = 1; s < M_; ++s) {
    const float px = cx[last], py = cy[last], pz = cz[last];
    float bv = -1.0f; int bi = 0;
#pragma unroll
    for (int k = 0; k < N_ / 1024; ++k) {
      int i = tid + k * 1024;
      float dx = cx[i] - px, dy = cy[i] - py, dz = cz[i] - pz;
      float d = dx * dx + dy * dy + dz * dz;
      float nd = fminf(db[i], d);
      db[i] = nd;
      if (nd > bv || (nd == bv && i < bi)) { bv = nd; bi = i; }
    }
#pragma unroll
    for (int off = 16; off > 0; off >>= 1) {        // wave32 argmax
      float ov = __shfl_xor(bv, off, 32);
      int   oi = __shfl_xor(bi, off, 32);
      if (ov > bv || (ov == bv && oi < bi)) { bv = ov; bi = oi; }
    }
    if (lane == 0) { swv[wv] = bv; swi[wv] = bi; }
    __syncthreads();
    if (wv == 0) {
      float v2 = swv[lane]; int i2 = swi[lane];
#pragma unroll
      for (int off = 16; off > 0; off >>= 1) {
        float ov = __shfl_xor(v2, off, 32);
        int   oi = __shfl_xor(i2, off, 32);
        if (ov > v2 || (ov == v2 && oi < i2)) { v2 = ov; i2 = oi; }
      }
      if (lane == 0) swinner = i2;
    }
    __syncthreads();
    last = swinner;
    if (tid == 0) fps_idx[b * M_ + s] = last;
  }
}

// ---------------------------------------------------------------------------
// 2) Gather fps coords -> output (B,3,M)
// ---------------------------------------------------------------------------
__global__ void fps_gather_kernel(const float* __restrict__ coords,
                                  const int* __restrict__ fps_idx,
                                  float* __restrict__ fps_out) {
  int t = blockIdx.x * blockDim.x + threadIdx.x;
  if (t >= B_ * 3 * M_) return;
  int b = t / (3 * M_);
  int rem = t % (3 * M_);
  int d = rem / M_;
  int m = rem % M_;
  fps_out[t] = coords[(size_t)(b * 3 + d) * N_ + fps_idx[b * M_ + m]];
}

// ---------------------------------------------------------------------------
// 3) kNN: one wave32 per center. Per-lane register-sorted top-16 (static
//    index swap-insert), then 16-round wave merge via shfl_xor with per-lane
//    sorted lists parked in LDS. Emits local_coords = fps - knn.
// ---------------------------------------------------------------------------
__global__ void __launch_bounds__(256)
knn_kernel(const float* __restrict__ coords, const int* __restrict__ fps_idx,
           int* __restrict__ knn_idx, float* __restrict__ local_out) {
  const int lane = threadIdx.x & 31;
  const int wv   = threadIdx.x >> 5;
  const int gw   = blockIdx.x * 8 + wv;          // grid is exact: B_*M_ waves
  const int b = gw / M_, m = gw % M_;
  __shared__ float sd[8][512];
  __shared__ int   si[8][512];
  const float* px = coords + (size_t)(b * 3 + 0) * N_;
  const float* py = coords + (size_t)(b * 3 + 1) * N_;
  const float* pz = coords + (size_t)(b * 3 + 2) * N_;
  const int ci = fps_idx[b * M_ + m];
  const float fx = px[ci], fy = py[ci], fz = pz[ci];
  float bd[KNN]; int bi[KNN];
#pragma unroll
  for (int p = 0; p < KNN; ++p) { bd[p] = 3.4e38f; bi[p] = 0; }
  for (int n = lane; n < N_; n += 32) {
    float dx = px[n] - fx, dy = py[n] - fy, dz = pz[n] - fz;
    float d = dx * dx + dy * dy + dz * dz;
    if (d < bd[KNN - 1]) {
      float cd = d; int cidx = n;
#pragma unroll
      for (int p = 0; p < KNN; ++p) {
        if (cd < bd[p]) {
          float td = bd[p]; int ti = bi[p];
          bd[p] = cd; bi[p] = cidx; cd = td; cidx = ti;
        }
      }
    }
  }
#pragma unroll
  for (int p = 0; p < KNN; ++p) {
    sd[wv][lane * 16 + p] = bd[p];
    si[wv][lane * 16 + p] = bi[p];
  }
  __syncthreads();
  int head = 0;
  for (int j = 0; j < KNN; ++j) {
    float v  = (head < KNN) ? sd[wv][lane * 16 + head] : 3.4e38f;
    int   id = (head < KNN) ? si[wv][lane * 16 + head] : 0;
    float mv = v; int mi = id; int ml = lane;
#pragma unroll
    for (int off = 16; off > 0; off >>= 1) {
      float ov = __shfl_xor(mv, off, 32);
      int   oi = __shfl_xor(mi, off, 32);
      int   ol = __shfl_xor(ml, off, 32);
      if (ov < mv || (ov == mv && ol < ml)) { mv = ov; mi = oi; ml = ol; }
    }
    if (ml == lane) head++;
    if (lane == j) {
      knn_idx[(size_t)gw * KNN + j] = mi;
      local_out[((size_t)(b * 3 + 0) * M_ + m) * KNN + j] = fx - px[mi];
      local_out[((size_t)(b * 3 + 1) * M_ + m) * KNN + j] = fy - py[mi];
      local_out[((size_t)(b * 3 + 2) * M_ + m) * KNN + j] = fz - pz[mi];
    }
  }
}

// ---------------------------------------------------------------------------
// 4) Convert W to f16 + zero BN accumulators (re-zeroed every call).
// ---------------------------------------------------------------------------
__global__ void prep_kernel(const float* __restrict__ W, _Float16* __restrict__ Wh,
                            float* __restrict__ sums, float* __restrict__ sqs) {
  int t = threadIdx.x;
  for (int i = t; i < COUT * CIN; i += blockDim.x) Wh[i] = (_Float16)W[i];
  if (t < COUT) { sums[t] = 0.f; sqs[t] = 0.f; }
}

// ---------------------------------------------------------------------------
// 5) Transpose x (B,64,N) f32 -> xT (B,N,64) f16 via LDS tile so the gather
//    in the GEMM is a contiguous 128B row copy.
// ---------------------------------------------------------------------------
__global__ void __launch_bounds__(256)
transpose_kernel(const float* __restrict__ x, _Float16* __restrict__ xT) {
  __shared__ _Float16 tile[64][72];
  const int blk = blockIdx.x;                    // B_*(N_/64)
  const int b  = blk / (N_ / 64);
  const int n0 = (blk % (N_ / 64)) * 64;
  const int tid = threadIdx.x;
  for (int e = tid; e < 64 * 64; e += 256) {
    int c = e >> 6, n = e & 63;
    tile[n][c] = (_Float16)x[((size_t)(b * CIN + c)) * N_ + n0 + n];
  }
  __syncthreads();
  int row = tid >> 2, ch0 = (tid & 3) * 16;
  _Float16* dst = xT + ((size_t)b * N_ + n0 + row) * CIN + ch0;
#pragma unroll
  for (int i = 0; i < 16; ++i) dst[i] = tile[row][ch0 + i];
}

// ---------------------------------------------------------------------------
// 6) WMMA GEMM with fused gather: h(128 x S) = W(128x64) * X(64 x S).
//    Block = 256 threads = 8 waves = all 128 out channels, owns 256 columns.
//    W staged in LDS once; A fragments hoisted (invariant over tiles).
//    X tiles (16 cols x 64 ch f16 = 2KB) gathered from L2-resident xT via
//    global_load_async_to_lds_b128 into double-buffered LDS, pipelined:
//    issue tile t+1 -> wmma tile t -> s_wait_asynccnt 0 -> barrier.
//    BN sum/sumsq accumulated in registers over all tiles, one width-16
//    shfl reduction + 8 atomics per wave at the end.
// ---------------------------------------------------------------------------
__global__ void __launch_bounds__(256)
gemm_kernel(const _Float16* __restrict__ Wh, const _Float16* __restrict__ xT,
            const int* __restrict__ knn_idx, float* __restrict__ hout,
            float* __restrict__ sums, float* __restrict__ sqs) {
  __shared__ __align__(32) _Float16 sW[COUT * CIN];   // 16 KB
  __shared__ __align__(32) _Float16 sX[2][16 * CIN];  // 2 x 2 KB
  const int tid = threadIdx.x;
  const int sblk = blockIdx.x * 256;                  // 256 columns per block
  const int b = sblk / (M_ * KNN);                    // uniform per block
  {
    uint4* dW = (uint4*)sW; const uint4* gW = (const uint4*)Wh;
#pragma unroll
    for (int i = 0; i < 4; ++i) dW[tid + i * 256] = gW[tid + i * 256];
  }
  const int col = tid >> 3, chunk = tid & 7;          // loader roles (tid<128)
  if (tid < 128) {
    int idx = knn_idx[sblk + col];
    const uint4* src = (const uint4*)(xT + ((size_t)b * N_ + idx) * CIN) + chunk;
    unsigned lds = (unsigned)(size_t)&((uint4*)sX[0])[col * 8 + chunk];
    asm volatile("global_load_async_to_lds_b128 %0, %1, off"
                 :: "v"(lds), "v"((unsigned long long)src) : "memory");
  }
  asm volatile("s_wait_asynccnt 0x0" ::: "memory");
  __syncthreads();

  const int lane = tid & 31, wv = tid >> 5;
  const int row  = wv * 16 + (lane & 15);
  const int offA = (lane < 16) ? 0 : 8;
  const int offB = (lane < 16) ? 0 : 16;
  const int ncol = lane & 15;

  // A fragments (16x32 f16, contiguous k = 16*(lane/16) + s) for K-chunks 0..31 and 32..63
  v16h a0, a1;
  {
    const v8h* p0 = (const v8h*)(sW + row * CIN + offB);
    a0 = __builtin_shufflevector(p0[0], p0[1], 0, 1, 2, 3, 4, 5, 6, 7,
                                 8, 9, 10, 11, 12, 13, 14, 15);
    const v8h* p1 = (const v8h*)(sW + row * CIN + 32 + offB);
    a1 = __builtin_shufflevector(p1[0], p1[1], 0, 1, 2, 3, 4, 5, 6, 7,
                                 8, 9, 10, 11, 12, 13, 14, 15);
  }

  float asum[8], asq[8];
#pragma unroll
  for (int r = 0; r < 8; ++r) { asum[r] = 0.f; asq[r] = 0.f; }
  const size_t MK = (size_t)M_ * KNN;
  float* hb0 = hout + (size_t)b * COUT * MK + (size_t)(sblk % (int)MK);

#pragma unroll 2
  for (int t = 0; t < 16; ++t) {
    const int cur = t & 1;
    if (t < 15 && tid < 128) {                       // prefetch next tile
      int idx = knn_idx[sblk + (t + 1) * 16 + col];
      const uint4* src = (const uint4*)(xT + ((size_t)b * N_ + idx) * CIN) + chunk;
      unsigned lds = (unsigned)(size_t)&((uint4*)sX[cur ^ 1])[col * 8 + chunk];
      asm volatile("global_load_async_to_lds_b128 %0, %1, off"
                   :: "v"(lds), "v"((unsigned long long)src) : "memory");
    }
    const _Float16* xb = sX[cur] + ncol * CIN + offB;
    v16h b0 = *(const v16h*)(xb);
    v16h b1 = *(const v16h*)(xb + 32);
    v8f acc = {};
    acc = __builtin_amdgcn_wmma_f32_16x16x32_f16(
        false, a0, false, b0, (short)0, acc, false, false);
    acc = __builtin_amdgcn_wmma_f32_16x16x32_f16(
        false, a1, false, b1, (short)0, acc, false, false);
    float* hb = hb0 + t * 16 + ncol;
#pragma unroll
    for (int r = 0; r < 8; ++r) {
      float v = acc[r];
      hb[(size_t)(wv * 16 + r + offA) * MK] = v;     // 64B coalesced segments
      asum[r] += v;
      asq[r]  += v * v;
    }
    asm volatile("s_wait_asynccnt 0x0" ::: "memory");
    __syncthreads();
  }
#pragma unroll
  for (int r = 0; r < 8; ++r) {
    float sv = asum[r], sq = asq[r];
#pragma unroll
    for (int off = 1; off < 16; off <<= 1) {         // half-wave reduce
      sv += __shfl_xor(sv, off, 16);
      sq += __shfl_xor(sq, off, 16);
    }
    if ((lane & 15) == 0) {
      const int c = wv * 16 + r + offA;
      atomicAdd(&sums[c], sv);
      atomicAdd(&sqs[c], sq);
    }
  }
}

// ---------------------------------------------------------------------------
// 7) BN stats -> mean / inv_std
// ---------------------------------------------------------------------------
__global__ void bn_finalize_kernel(const float* __restrict__ sums,
                                   const float* __restrict__ sqs,
                                   float* __restrict__ meanb,
                                   float* __restrict__ invb) {
  int c = threadIdx.x;
  if (c < COUT) {
    float mu  = sums[c] * (1.0f / (float)S_);
    float var = sqs[c] * (1.0f / (float)S_) - mu * mu;
    meanb[c] = mu;
    invb[c]  = rsqrtf(var + BN_EPS);
  }
}

// ---------------------------------------------------------------------------
// 8) Normalize + ReLU in place (knn_mlp_x output) + max over k -> y
// ---------------------------------------------------------------------------
__global__ void __launch_bounds__(256)
bn_apply_kernel(float* __restrict__ h, const float* __restrict__ meanb,
                const float* __restrict__ invb, const float* __restrict__ gamma,
                const float* __restrict__ beta, float* __restrict__ y) {
  int t = blockIdx.x * blockDim.x + threadIdx.x;   // B_*COUT*M_ threads exact
  int c = (t / M_) % COUT;
  float mu = meanb[c];
  float iv = invb[c] * gamma[c];
  float bt = beta[c];
  float4* p = (float4*)(h + (size_t)t * KNN);
  float mx = -3.4e38f;
#pragma unroll
  for (int i = 0; i < 4; ++i) {
    float4 v = p[i];
    v.x = fmaxf((v.x - mu) * iv + bt, 0.f);
    v.y = fmaxf((v.y - mu) * iv + bt, 0.f);
    v.z = fmaxf((v.z - mu) * iv + bt, 0.f);
    v.w = fmaxf((v.w - mu) * iv + bt, 0.f);
    mx = fmaxf(mx, fmaxf(fmaxf(v.x, v.y), fmaxf(v.z, v.w)));
    p[i] = v;
  }
  y[t] = mx;
}

// ---------------------------------------------------------------------------
// Host launcher
// ---------------------------------------------------------------------------
extern "C" void kernel_launch(void* const* d_in, const int* in_sizes, int n_in,
                              void* d_out, int out_size, void* d_ws, size_t ws_size,
                              hipStream_t stream) {
  (void)in_sizes; (void)n_in; (void)out_size; (void)ws_size;
  const float* x      = (const float*)d_in[0];   // (B,64,N)
  const float* coords = (const float*)d_in[1];   // (B,3,N)
  const float* W      = (const float*)d_in[2];   // (128,64)
  const float* gamma  = (const float*)d_in[3];
  const float* beta   = (const float*)d_in[4];

  float* out = (float*)d_out;
  // tuple: y (B,128,M) | fps_coords (B,3,M) | knn_mlp_x (B,128,M,K) | local_coords (B,3,M,K)
  float* y_out     = out;
  float* fps_out   = out + (size_t)B_ * COUT * M_;
  float* h_out     = fps_out + (size_t)B_ * 3 * M_;
  float* local_out = h_out + (size_t)B_ * COUT * M_ * KNN;

  char* ws = (char*)d_ws;
  float*    dist    = (float*)(ws + 0);          // B*N f32   = 256 KB
  int*      fps_idx = (int*)  (ws + 262144);     // B*M i32   =  64 KB
  int*      knn_idx = (int*)  (ws + 327680);     // B*M*K i32 =   1 MB
  _Float16* Wh      = (_Float16*)(ws + 1376256); // 128*64 f16=  16 KB
  float*    sums    = (float*)(ws + 1392640);    // 128 f32
  float*    sqs     = (float*)(ws + 1393152);    // 128 f32
  float*    meanb   = (float*)(ws + 1393664);    // 128 f32
  float*    invb    = (float*)(ws + 1394176);    // 128 f32
  _Float16* xT      = (_Float16*)(ws + 1394688); // B*N*64 f16 =  8 MB

  fps_kernel<<<B_, 1024, 0, stream>>>(coords, dist, fps_idx);
  fps_gather_kernel<<<(B_ * 3 * M_ + 255) / 256, 256, 0, stream>>>(coords, fps_idx, fps_out);
  knn_kernel<<<(B_ * M_) / 8, 256, 0, stream>>>(coords, fps_idx, knn_idx, local_out);
  prep_kernel<<<1, 256, 0, stream>>>(W, Wh, sums, sqs);
  transpose_kernel<<<B_ * (N_ / 64), 256, 0, stream>>>(x, xT);
  gemm_kernel<<<S_ / 256, 256, 0, stream>>>(Wh, xT, knn_idx, h_out, sums, sqs);
  bn_finalize_kernel<<<1, COUT, 0, stream>>>(sums, sqs, meanb, invb);
  bn_apply_kernel<<<(B_ * COUT * M_) / 256, 256, 0, stream>>>(h_out, meanb, invb, gamma, beta, y_out);
}
